// CustomAttentionLayer_13563506720864
// MI455X (gfx1250) — matrix operand order repaired and checked by reading, not compiled, not run on hardware
//
#include <hip/hip_runtime.h>
#include <hip/hip_fp16.h>

typedef __attribute__((ext_vector_type(16))) _Float16 v16h;
typedef __attribute__((ext_vector_type(8)))  float    v8f;

#define CDIM 256
#define NEG_SLOPE 0.2f

// ---------------------------------------------------------------------------
// float <-> order-preserving uint encoding (for atomicMax over signed floats)
// ---------------------------------------------------------------------------
__device__ __forceinline__ unsigned f32_to_ord(float f) {
  unsigned u = __float_as_uint(f);
  return (u & 0x80000000u) ? ~u : (u | 0x80000000u);
}
__device__ __forceinline__ float ord_to_f32(unsigned u) {
  return (u & 0x80000000u) ? __uint_as_float(u ^ 0x80000000u)
                           : __uint_as_float(~u);
}

// ---------------------------------------------------------------------------
// Phase 0: init segment-max (encoded -inf == 0) and segment-sum buffers
// ---------------------------------------------------------------------------
__global__ void gat_init_kernel(unsigned* __restrict__ m_bits,
                                float* __restrict__ denom, int n) {
  int i = blockIdx.x * blockDim.x + threadIdx.x;
  if (i < n) { m_bits[i] = 0u; denom[i] = 0.0f; }
}

// ---------------------------------------------------------------------------
// Phase 1: WMMA GEMV. Each wave handles a 16-row tile of x and computes both
// s_src (D column N=0) and s_dst (N=1) via v_wmma_f32_16x16x32_f16, K=256.
// 256 threads/block = 8 waves = 8 tiles per block.
// ---------------------------------------------------------------------------
__global__ void gat_score_wmma_kernel(const float* __restrict__ x,
                                      const float* __restrict__ att,
                                      float* __restrict__ s_src,
                                      float* __restrict__ s_dst, int n) {
  const int lane = threadIdx.x & 31;
  const int wave = threadIdx.x >> 5;
  const int tile = blockIdx.x * 8 + wave;
  const int rowBase = tile * 16;
  if (rowBase >= n) return;                 // wave-uniform exit

  const int mcol = lane & 15;               // A: M row | B/D: N column
  const int kh   = lane >> 4;               // lane-half -> K sub-block
  int row = rowBase + mcol;
  if (row >= n) row = n - 1;                // branchless clamp, EXEC stays full
  const float* __restrict__ xrow = x + (size_t)row * CDIM;
  // B column 0 = att[0:256] (src), column 1 = att[256:512] (dst), else zero.
  // All lanes load from valid att memory; inactive columns are zeroed in
  // registers (no divergent loads, no EXEC churn).
  const float* __restrict__ w = (mcol == 0) ? att : (att + CDIM);
  const float colMask = (mcol < 2) ? 1.0f : 0.0f;

  v8f acc = {};
#pragma unroll
  for (int k0 = 0; k0 < CDIM; k0 += 32) {
    // A-matrix 16x32 f16 layout (ISA 7.12.2):
    //   elems 0..7  -> K = k0 + 8*kh + i
    //   elems 8..15 -> K = k0 + 16 + 8*kh + (i-8)
    v16h a;
#pragma unroll
    for (int i = 0; i < 8; ++i)
      a[i] = (_Float16)xrow[k0 + 8 * kh + i];
#pragma unroll
    for (int i = 0; i < 8; ++i)
      a[8 + i] = (_Float16)xrow[k0 + 16 + 8 * kh + i];

    // B-matrix 32x16 f16: lane = N column, 16 elems = K (8*kh + (i&7) + 16*(i>>3))
    v16h b;
#pragma unroll
    for (int i = 0; i < 16; ++i) {
      float v = w[k0 + 8 * kh + (i & 7) + 16 * (i >> 3)];        // unconditional, contiguous
      b[i] = (_Float16)(v * colMask);       // branchless zeroing
    }
    acc = __builtin_amdgcn_wmma_f32_16x16x32_f16(
        /*neg_a=*/false, a, /*neg_b=*/false, b,
        /*c_mod=*/(short)0, acc, /*reuse_a=*/false, /*reuse_b=*/false);
  }

  // D layout: lane l holds column N=l%16, rows M = 8*(l/16) + r (r=vgpr 0..7)
  const int mBase = rowBase + 8 * kh;
  if (mcol == 0) {
#pragma unroll
    for (int r = 0; r < 8; ++r) {
      int rr = mBase + r;
      if (rr < n) s_src[rr] = acc[r];
    }
  } else if (mcol == 1) {
#pragma unroll
    for (int r = 0; r < 8; ++r) {
      int rr = mBase + r;
      if (rr < n) s_dst[rr] = acc[r];
    }
  }
}

// ---------------------------------------------------------------------------
// Phase 2: alpha = leaky_relu(s_src[row] + s_dst[col]); stash alpha in d_out;
//          segment max over row via encoded atomicMax (s_* are L2-resident)
// ---------------------------------------------------------------------------
__global__ void gat_alpha_max_kernel(const int* __restrict__ row,
                                     const int* __restrict__ col,
                                     const float* __restrict__ s_src,
                                     const float* __restrict__ s_dst,
                                     float* __restrict__ alpha_out,
                                     unsigned* __restrict__ m_bits, int nE) {
  int e = blockIdx.x * blockDim.x + threadIdx.x;
  if (e >= nE) return;
  int r = row[e], c = col[e];
  float a = s_src[r] + s_dst[c];
  a = (a > 0.0f) ? a : NEG_SLOPE * a;
  alpha_out[e] = a;
  atomicMax(&m_bits[r], f32_to_ord(a));
}

// ---------------------------------------------------------------------------
// Phase 3: e = exp(alpha - m[row]); overwrite d_out with e; segment-sum denom
// ---------------------------------------------------------------------------
__global__ void gat_exp_sum_kernel(const int* __restrict__ row,
                                   float* __restrict__ buf,
                                   const unsigned* __restrict__ m_bits,
                                   float* __restrict__ denom, int nE) {
  int e = blockIdx.x * blockDim.x + threadIdx.x;
  if (e >= nE) return;
  int r = row[e];
  float mv = ord_to_f32(m_bits[r]);
  float ev = __expf(buf[e] - mv);           // argument <= 0
  buf[e] = ev;
  atomicAdd(&denom[r], ev);
}

// ---------------------------------------------------------------------------
// Phase 4: out = e / denom[row]
// ---------------------------------------------------------------------------
__global__ void gat_norm_kernel(const int* __restrict__ row,
                                float* __restrict__ buf,
                                const float* __restrict__ denom, int nE) {
  int e = blockIdx.x * blockDim.x + threadIdx.x;
  if (e >= nE) return;
  buf[e] = buf[e] / denom[row[e]];
}

// ---------------------------------------------------------------------------
extern "C" void kernel_launch(void* const* d_in, const int* in_sizes, int n_in,
                              void* d_out, int out_size, void* d_ws, size_t ws_size,
                              hipStream_t stream) {
  const float* x   = (const float*)d_in[0];
  const float* att = (const float*)d_in[1];
  const int*   ei  = (const int*)d_in[2];

  const int n  = in_sizes[0] / CDIM;   // 100000
  const int nE = in_sizes[2] / 2;      // 3200000
  const int* row = ei;                 // edge_index[0]
  const int* col = ei + nE;            // edge_index[1]
  float* out = (float*)d_out;          // (1, E) flat

  // workspace: s_src[n] | s_dst[n] | m_bits[n] | denom[n]  (4n floats)
  float*    s_src  = (float*)d_ws;
  float*    s_dst  = s_src + n;
  unsigned* m_bits = (unsigned*)(s_dst + n);
  float*    denom  = (float*)(m_bits + n);

  const int T = 256;
  gat_init_kernel<<<(n + T - 1) / T, T, 0, stream>>>(m_bits, denom, n);

  const int tiles = (n + 15) / 16;                 // 16 rows per wave
  const int blocks = (tiles + 7) / 8;              // 8 waves per block
  gat_score_wmma_kernel<<<blocks, 256, 0, stream>>>(x, att, s_src, s_dst, n);

  gat_alpha_max_kernel<<<(nE + T - 1) / T, T, 0, stream>>>(row, col, s_src,
                                                           s_dst, out, m_bits, nE);
  gat_exp_sum_kernel<<<(nE + T - 1) / T, T, 0, stream>>>(row, out, m_bits,
                                                         denom, nE);
  gat_norm_kernel<<<(nE + T - 1) / T, T, 0, stream>>>(row, out, denom, nE);
}
